// ContrastiveLearningLoss_41884521071006
// MI455X (gfx1250) — matrix operand order repaired and checked by reading, not compiled, not run on hardware
//
#include <hip/hip_runtime.h>
#include <hip/hip_bf16.h>
#include <stdint.h>

typedef __attribute__((ext_vector_type(16))) _Float16 v16h;
typedef __attribute__((ext_vector_type(8)))  _Float16 v8h;
typedef __attribute__((ext_vector_type(4)))  _Float16 v4h;
typedef __attribute__((ext_vector_type(8)))  float    v8f;

#define E_EDGES   8192
#define DIM       256
#define TAU_INV   10.0f          // 1 / TEMPERATURE
#define CSPLIT    8              // column splits (gridDim.y)
#define TILES_N   (E_EDGES / 16) // 512 column tiles
#define TILES_PER_SPLIT (TILES_N / CSPLIT)
#define TILE_HALVES 4096         // 16 cols x 256 K values (8 KB)

// ---------------------------------------------------------------------------
// CDNA5 async global->LDS copy (ASYNCcnt-tracked, bypasses VGPR return path)
// ---------------------------------------------------------------------------
__device__ __forceinline__ void async_b128(uint32_t lds_off, const _Float16* g)
{
    asm volatile("global_load_async_to_lds_b128 %0, %1, off"
                 :: "v"(lds_off), "v"(g) : "memory");
}
__device__ __forceinline__ void wait_async0()
{
    asm volatile("s_wait_asynccnt 0x0" ::: "memory");
}

// ---------------------------------------------------------------------------
// Kernel 1: fp32 -> f16 conversion of the embedding matrix (L2-resident copy)
// ---------------------------------------------------------------------------
__global__ __launch_bounds__(256)
void cvt_f32_to_f16(const float* __restrict__ src, _Float16* __restrict__ dst, int n)
{
    int i4 = (blockIdx.x * blockDim.x + threadIdx.x) * 4;
    if (i4 < n) {
        float4 v = *(const float4*)(src + i4);
        v4h h;
        h[0] = (_Float16)v.x; h[1] = (_Float16)v.y;
        h[2] = (_Float16)v.z; h[3] = (_Float16)v.w;
        *(v4h*)(dst + i4) = h;
    }
}

// ---------------------------------------------------------------------------
// Kernel 2: fused tiled A*A^T (WMMA f16 -> f32) + mask + exp + row partials.
// B tiles staged once per block into double-buffered LDS via async copies.
// ---------------------------------------------------------------------------
__global__ __launch_bounds__(256)
void contrastive_tiles(const _Float16* __restrict__ Ah,
                       const float*    __restrict__ scores,
                       const int*      __restrict__ eidx,
                       float* __restrict__ posPart,
                       float* __restrict__ negPart)
{
    __shared__ __align__(32) _Float16 tile[2][TILE_HALVES];   // 2 x 8 KB

    const int t    = threadIdx.x;
    const int lane = t & 31;
    const int wave = t >> 5;
    const int rowTile = blockIdx.x * 8 + wave;   // 0..511
    const int i0 = rowTile * 16;
    const int g   = lane >> 4;                   // half-wave group (0/1)
    const int l15 = lane & 15;

    // ---- staging map: thread t copies LDS bytes [t*32, t*32+32) of a tile ----
    // dest t*32 = kt2*1024 + g2*512 + c*32  with kt2=t>>5, g2=(t>>4)&1, c=t&15
    // source    = Ah[(j0+c)*256 + kt2*32 + g2*16 .. +15]   (two 16B chunks)
    const int c   = t & 15;
    const int g2  = (t >> 4) & 1;
    const int kt2 = t >> 5;
    const int ct0 = blockIdx.y * TILES_PER_SPLIT;
    const _Float16* src = Ah + (size_t)(ct0 * 16 + c) * DIM + kt2 * 32 + g2 * 16;
    const uint32_t dstBuf0 = (uint32_t)(uintptr_t)(&tile[0][0]) + (uint32_t)t * 32u;
    const uint32_t dstBuf1 = (uint32_t)(uintptr_t)(&tile[1][0]) + (uint32_t)t * 32u;

    // ---- A fragments: 16x32 f16 per K-step, contiguous k = 16*g + s (reg-resident) ----
    const int m     = i0 + l15;
    const int koffA = g * 16;
    v16h afrag[8];
#pragma unroll
    for (int kt = 0; kt < 8; ++kt) {
        const _Float16* pa = Ah + (size_t)m * DIM + kt * 32 + koffA;
        v8h lo = *(const v8h*)(pa);
        v8h hi = *(const v8h*)(pa + 8);
#pragma unroll
        for (int q = 0; q < 8; ++q) { afrag[kt][q] = lo[q]; afrag[kt][q + 8] = hi[q]; }
    }

    // ---- per-row (i side) scalars: element e -> row i0+8g+e ----
    float si[8]; int ri[8], ci[8], iv[8];
#pragma unroll
    for (int e = 0; e < 8; ++e) {
        int i  = i0 + g * 8 + e;
        iv[e] = i;
        si[e] = scores[i];
        ri[e] = eidx[i];
        ci[e] = eidx[E_EDGES + i];
    }

    float pacc[8], nacc[8];
#pragma unroll
    for (int e = 0; e < 8; ++e) { pacc[e] = 0.0f; nacc[e] = 0.0f; }

    // ---- prologue: stage first tile into buffer 0 ----
    async_b128(dstBuf0,       src);
    async_b128(dstBuf0 + 16u, src + 8);
    src += 16 * DIM;                       // advance one column tile (8 KB)
    wait_async0();
    __syncthreads();

    for (int it = 0; it < TILES_PER_SPLIT; ++it) {
        const int ct = ct0 + it;
        const int cur = it & 1;

        // stage next tile into the other buffer (overlaps with compute below)
        if (it + 1 < TILES_PER_SPLIT) {
            uint32_t d = cur ? dstBuf0 : dstBuf1;
            async_b128(d,       src);
            async_b128(d + 16u, src + 8);
            src += 16 * DIM;
        }

        const int j0 = ct * 16;
        const int jl = j0 + l15;                 // this lane's output column
        const float sj = scores[jl];
        const int rj = eidx[jl];
        const int cj = eidx[E_EDGES + jl];

        // B fragments from LDS: lane-linear, conflict-free (lane*32 bytes)
        const _Float16* tb = &tile[cur][0];
        v16h bfr[8];
#pragma unroll
        for (int kt = 0; kt < 8; ++kt)
            bfr[kt] = *(const v16h*)(tb + kt * 512 + lane * 16);

        v8f acc = {};
#pragma unroll
        for (int kt = 0; kt < 8; ++kt)
            acc = __builtin_amdgcn_wmma_f32_16x16x32_f16(
                false, afrag[kt], false, bfr[kt], (short)0, acc, false, false);

        // fused epilogue: C/D layout -> element e is (row iv[e], col jl)
#pragma unroll
        for (int e = 0; e < 8; ++e) {
            float sim = acc[e] * TAU_INV;
            bool shared = (((ri[e] == rj) | (ri[e] == cj) |
                            (ci[e] == rj) | (ci[e] == cj)) != 0) && (iv[e] != jl);
            bool p = (__builtin_fabsf(si[e] - sj) < 0.1f) || shared;
            pacc[e] += p ? sim : 0.0f;
            nacc[e] += p ? 0.0f : __expf(sim);
        }

        // my async stage is done; everyone done reading current buffer
        wait_async0();
        __syncthreads();
    }

    // reduce each half-wave's 16 lanes (16 columns per tile) -> per-row sums
#pragma unroll
    for (int e = 0; e < 8; ++e) {
#pragma unroll
        for (int msk = 1; msk < 16; msk <<= 1) {
            pacc[e] += __shfl_xor(pacc[e], msk, 32);
            nacc[e] += __shfl_xor(nacc[e], msk, 32);
        }
    }
    if (l15 == 0) {
#pragma unroll
        for (int e = 0; e < 8; ++e) {
            int row = i0 + g * 8 + e;
            posPart[(size_t)blockIdx.y * E_EDGES + row] = pacc[e];
            negPart[(size_t)blockIdx.y * E_EDGES + row] = nacc[e];
        }
    }
}

// ---------------------------------------------------------------------------
// Kernel 3: deterministic finalize — fixed-order sum of column splits,
// per-row loss, block reduction to the scalar mean.
// ---------------------------------------------------------------------------
__global__ __launch_bounds__(256)
void finalize_loss(const float* __restrict__ posPart,
                   const float* __restrict__ negPart,
                   float* __restrict__ out)
{
    __shared__ float red[256];
    int t = threadIdx.x;
    float local = 0.0f;
    for (int r = t; r < E_EDGES; r += 256) {
        float p = 0.0f, n = 0.0f;
#pragma unroll
        for (int s = 0; s < CSPLIT; ++s) {
            p += posPart[(size_t)s * E_EDGES + r];
            n += negPart[(size_t)s * E_EDGES + r];
        }
        local += -__logf(p / (p + n + 1e-8f));
    }
    red[t] = local;
    __syncthreads();
    for (int off = 128; off > 0; off >>= 1) {
        if (t < off) red[t] += red[t + off];
        __syncthreads();
    }
    if (t == 0) out[0] = red[0] * (1.0f / (float)E_EDGES);
}

// ---------------------------------------------------------------------------
extern "C" void kernel_launch(void* const* d_in, const int* in_sizes, int n_in,
                              void* d_out, int out_size, void* d_ws, size_t ws_size,
                              hipStream_t stream)
{
    const float* emb    = (const float*)d_in[0];   // [8192, 256] fp32
    const int*   eidx   = (const int*)  d_in[1];   // [2, 8192]
    const float* scores = (const float*)d_in[2];   // [8192]
    float* out = (float*)d_out;

    // workspace layout
    _Float16* Ah = (_Float16*)d_ws;                                   // 4 MB
    float* posPart = (float*)((char*)d_ws + (size_t)E_EDGES * DIM * sizeof(_Float16));
    float* negPart = posPart + (size_t)CSPLIT * E_EDGES;              // 256 KB each

    // 1) convert embeddings to f16 (vectorized, 4 elems/thread)
    int n = E_EDGES * DIM;
    cvt_f32_to_f16<<<n / 4 / 256, 256, 0, stream>>>(emb, Ah, n);

    // 2) fused WMMA tiles: 64 row-groups x 8 column splits, 8 waves/block
    dim3 grid(E_EDGES / 128, CSPLIT);
    contrastive_tiles<<<grid, 256, 0, stream>>>(Ah, scores, eidx, posPart, negPart);

    // 3) deterministic reduction to the scalar loss
    finalize_loss<<<1, 256, 0, stream>>>(posPart, negPart, out);
}
